// MultiHeadAttention_39719857553421
// MI455X (gfx1250) — matrix operand order repaired and checked by reading, not compiled, not run on hardware
//
#include <hip/hip_runtime.h>

typedef __attribute__((ext_vector_type(16))) __bf16 bf16x16;
typedef __attribute__((ext_vector_type(8)))  float  f32x8;

struct U4 { unsigned int x, y, z, w; };

union Frag {
    bf16x16      v;
    U4           q[2];
    unsigned int u[8];
};

// Native f32 -> bf16 (compiler emits v_cvt_pk_bf16_f32 / native cvt when the
// target supports it; RNE software sequence otherwise).
__device__ __forceinline__ unsigned short f2bf(float x) {
    __bf16 t = (__bf16)x;
    return __builtin_bit_cast(unsigned short, t);
}
__device__ __forceinline__ unsigned int pack2(float lo, float hi) {
    return (unsigned int)f2bf(lo) | ((unsigned int)f2bf(hi) << 16);
}

// ---------------------------------------------------------------------------
// GEMM: OUT[M,N] = X[M,K] @ W[N,K]^T + bias[N] (+ R[M,N]),  M=4096,N=1024,K=1024
// 128x64 tile / 256 threads (8 waves). LDS double-buffered: next tile's global
// loads are issued before the current tile's 4 WMMAs; one barrier per k-step.
// ---------------------------------------------------------------------------
template <bool HAS_R>
__launch_bounds__(256)
__global__ void gemm_xwt_bias(const float* __restrict__ X,
                              const float* __restrict__ W,
                              const float* __restrict__ bias,
                              const float* __restrict__ R,
                              float* __restrict__ OUT)
{
    __shared__ unsigned int sA[2][128 * 16];  // 128 rows x 32 bf16 each
    __shared__ unsigned int sB[2][64 * 16];   // 64 rows x 32 bf16 each

    const int tid  = threadIdx.x;
    const int lane = tid & 31;
    const int w    = tid >> 5;
    const int hl   = lane >> 4;
    const int l16  = lane & 15;
    const int m0   = blockIdx.x * 128;
    const int n0   = blockIdx.y * 64;

    f32x8 acc[4] = {};

    const int arow = tid >> 1;           // 0..127
    const int asl  = (tid & 1) * 8;      // dword slot in LDS row
    const int brow = tid >> 2;           // 0..63
    const int bsl  = (tid & 3) * 4;

    const float* pax = X + (size_t)(m0 + arow) * 1024 + (tid & 1) * 16;
    const float* pbx = W + (size_t)(n0 + brow) * 1024 + (tid & 3) * 8;

    // preload tile 0
    {
        const float4* pa = (const float4*)pax;
        float4 a0 = pa[0], a1 = pa[1], a2 = pa[2], a3 = pa[3];
        unsigned int* d = sA[0] + arow * 16 + asl;
        d[0] = pack2(a0.x, a0.y); d[1] = pack2(a0.z, a0.w);
        d[2] = pack2(a1.x, a1.y); d[3] = pack2(a1.z, a1.w);
        d[4] = pack2(a2.x, a2.y); d[5] = pack2(a2.z, a2.w);
        d[6] = pack2(a3.x, a3.y); d[7] = pack2(a3.z, a3.w);
        const float4* pb = (const float4*)pbx;
        float4 b0 = pb[0], b1 = pb[1];
        unsigned int* e = sB[0] + brow * 16 + bsl;
        e[0] = pack2(b0.x, b0.y); e[1] = pack2(b0.z, b0.w);
        e[2] = pack2(b1.x, b1.y); e[3] = pack2(b1.z, b1.w);
    }
    __syncthreads();

    const int mr = w * 16 + l16;
    for (int t = 0; t < 32; ++t) {
        const int cur = t & 1;
        const bool more = (t < 31);
        float4 a0{}, a1{}, a2{}, a3{}, b0{}, b1{};
        if (more) {                                   // issue next tile's loads early
            const float4* pa = (const float4*)(pax + (t + 1) * 32);
            a0 = pa[0]; a1 = pa[1]; a2 = pa[2]; a3 = pa[3];
            const float4* pb = (const float4*)(pbx + (t + 1) * 32);
            b0 = pb[0]; b1 = pb[1];
        }

        Frag a;
        // A (16x32): VGPR0-3 = K[hl*8 .. +7], VGPR4-7 = K[16+hl*8 .. +7]
        a.q[0] = *(const U4*)(sA[cur] + mr * 16 + hl * 4);
        a.q[1] = *(const U4*)(sA[cur] + mr * 16 + 8 + hl * 4);
        Frag bf[4];
#pragma unroll
        for (int f = 0; f < 4; ++f) {
            const int nr = f * 16 + l16;
            // B (32x16): VGPR0-3 = K[hl*8 .. +7], VGPR4-7 = K[16+hl*8 .. +7]
            bf[f].q[0] = *(const U4*)(sB[cur] + nr * 16 + hl * 4);
            bf[f].q[1] = *(const U4*)(sB[cur] + nr * 16 + 8 + hl * 4);
        }
        // 4 identical consecutive WMMAs sharing A -> reuse_a hint on 1..3
        acc[0] = __builtin_amdgcn_wmma_f32_16x16x32_bf16(false, a.v, false, bf[0].v, (short)0, acc[0], false, false);
        acc[1] = __builtin_amdgcn_wmma_f32_16x16x32_bf16(false, a.v, false, bf[1].v, (short)0, acc[1], true,  false);
        acc[2] = __builtin_amdgcn_wmma_f32_16x16x32_bf16(false, a.v, false, bf[2].v, (short)0, acc[2], true,  false);
        acc[3] = __builtin_amdgcn_wmma_f32_16x16x32_bf16(false, a.v, false, bf[3].v, (short)0, acc[3], true,  false);

        if (more) {                                   // pack into alternate buffer
            unsigned int* d = sA[cur ^ 1] + arow * 16 + asl;
            d[0] = pack2(a0.x, a0.y); d[1] = pack2(a0.z, a0.w);
            d[2] = pack2(a1.x, a1.y); d[3] = pack2(a1.z, a1.w);
            d[4] = pack2(a2.x, a2.y); d[5] = pack2(a2.z, a2.w);
            d[6] = pack2(a3.x, a3.y); d[7] = pack2(a3.z, a3.w);
            unsigned int* e = sB[cur ^ 1] + brow * 16 + bsl;
            e[0] = pack2(b0.x, b0.y); e[1] = pack2(b0.z, b0.w);
            e[2] = pack2(b1.x, b1.y); e[3] = pack2(b1.z, b1.w);
        }
        __syncthreads();
    }

#pragma unroll
    for (int f = 0; f < 4; ++f) {
        const int ncol = n0 + f * 16 + l16;
        const float bv = bias[ncol];
#pragma unroll
        for (int v = 0; v < 8; ++v) {
            const int mrow = m0 + w * 16 + v + 8 * hl;
            const size_t o = (size_t)mrow * 1024 + ncol;
            if (HAS_R) OUT[o] = acc[f][v] + bv + R[o];
            else       OUT[o] = acc[f][v] + bv;
        }
    }
}

// ---------------------------------------------------------------------------
// Attention: per (b, h, 32-row strip). Scores strip lives in LDS (32x1040 f32),
// softmax + mask in LDS, attn written to HBM once, P@V via WMMA.
// K tiles and V chunks are LDS double-buffered (one barrier per step).
// ---------------------------------------------------------------------------
#define SSTR 1040

__launch_bounds__(256)
__global__ void attn_kernel(const float* __restrict__ qb,
                            const float* __restrict__ kb,
                            const float* __restrict__ vb,
                            const int*   __restrict__ mask,
                            float* __restrict__ attn,
                            float* __restrict__ ctx)
{
    extern __shared__ char smem[];
    float*        sS  = (float*)smem;                                   // 32 x 1040 f32
    unsigned int* sQ  = (unsigned int*)(smem + 32 * SSTR * 4);          // 32 x 64 bf16
    unsigned int* sKR = (unsigned int*)(smem + 32 * SSTR * 4 + 4096);   // 16KB: K double-buf / V double-buf
    float*        sMF = (float*)(smem + 32 * SSTR * 4 + 20480);         // 1024 f32 mask factors

    const int tid  = threadIdx.x;
    const int lane = tid & 31;
    const int w    = tid >> 5;
    const int hl   = lane >> 4;
    const int l16  = lane & 15;
    const int b    = blockIdx.z;
    const int h    = blockIdx.y;
    const int i0   = blockIdx.x * 32;
    const int msub = w & 1;      // 2 subtiles of 16 rows
    const int nsub = w >> 1;     // 4 subtiles of 16 cols

    // ---- Stage 1: stage Q strip (bf16) + mask factors ----
    {
        const int r = tid >> 3, d0 = (tid & 7) * 8;
        const float4* p = (const float4*)(qb + ((size_t)(b * 1024 + i0 + r)) * 1024 + h * 64 + d0);
        float4 q0 = p[0], q1 = p[1];
        unsigned int* dst = sQ + r * 32 + d0 / 2;
        dst[0] = pack2(q0.x, q0.y); dst[1] = pack2(q0.z, q0.w);
        dst[2] = pack2(q1.x, q1.y); dst[3] = pack2(q1.z, q1.w);
        for (int j = tid; j < 1024; j += 256)
            sMF[j] = 1.0f - (float)mask[b * 1024 + j];
    }

    // ---- Stage 2: scores strip = Q @ K^T * (1/8), K tile double-buffered ----
    const int kn  = tid >> 2, ksl = (tid & 3) * 8;    // coop: row, dword slot
    const float* kbase = kb + ((size_t)(b * 1024 + kn)) * 1024 + h * 64 + (tid & 3) * 16;
    {
        const float4* p = (const float4*)kbase;       // tile 0
        float4 k0 = p[0], k1 = p[1], k2 = p[2], k3 = p[3];
        unsigned int* dst = sKR + kn * 32 + ksl;
        dst[0] = pack2(k0.x, k0.y); dst[1] = pack2(k0.z, k0.w);
        dst[2] = pack2(k1.x, k1.y); dst[3] = pack2(k1.z, k1.w);
        dst[4] = pack2(k2.x, k2.y); dst[5] = pack2(k2.z, k2.w);
        dst[6] = pack2(k3.x, k3.y); dst[7] = pack2(k3.z, k3.w);
    }
    __syncthreads();

    const int mr = msub * 16 + l16;
    const int nr = nsub * 16 + l16;
    for (int jt = 0; jt < 16; ++jt) {
        const unsigned int* sK = sKR + (jt & 1) * 2048;
        const bool more = (jt < 15);
        float4 k0{}, k1{}, k2{}, k3{};
        if (more) {                                   // next K tile: rows += 64
            const float4* p = (const float4*)(kbase + (size_t)(jt + 1) * 65536);
            k0 = p[0]; k1 = p[1]; k2 = p[2]; k3 = p[3];
        }

        f32x8 c = {};
#pragma unroll
        for (int ks = 0; ks < 2; ++ks) {
            Frag a, bf;
            a.q[0]  = *(const U4*)(sQ + mr * 32 + ks * 16 + hl * 4);
            a.q[1]  = *(const U4*)(sQ + mr * 32 + ks * 16 + 8 + hl * 4);
            bf.q[0] = *(const U4*)(sK + nr * 32 + ks * 16 + hl * 4);
            bf.q[1] = *(const U4*)(sK + nr * 32 + ks * 16 + 8 + hl * 4);
            c = __builtin_amdgcn_wmma_f32_16x16x32_bf16(false, a.v, false, bf.v, (short)0, c, false, false);
        }
#pragma unroll
        for (int v = 0; v < 8; ++v) {
            const int mrow = msub * 16 + v + 8 * hl;
            sS[mrow * SSTR + jt * 64 + nr] = c[v] * 0.125f;
        }

        if (more) {
            unsigned int* dst = sKR + ((jt + 1) & 1) * 2048 + kn * 32 + ksl;
            dst[0] = pack2(k0.x, k0.y); dst[1] = pack2(k0.z, k0.w);
            dst[2] = pack2(k1.x, k1.y); dst[3] = pack2(k1.z, k1.w);
            dst[4] = pack2(k2.x, k2.y); dst[5] = pack2(k2.z, k2.w);
            dst[6] = pack2(k3.x, k3.y); dst[7] = pack2(k3.z, k3.w);
        }
        __syncthreads();
    }

    // ---- Stage 3: softmax (reference: softmax FIRST, then mask-multiply) ----
    for (int rr = 0; rr < 4; ++rr) {
        const int r = w * 4 + rr;
        float* row = sS + r * SSTR;
        float p[32];
        float mx = -3.4e38f;
#pragma unroll
        for (int j4 = 0; j4 < 8; ++j4) {
            float4 t = *(const float4*)(row + lane * 32 + j4 * 4);
            p[j4 * 4 + 0] = t.x; p[j4 * 4 + 1] = t.y; p[j4 * 4 + 2] = t.z; p[j4 * 4 + 3] = t.w;
            mx = fmaxf(mx, fmaxf(fmaxf(t.x, t.y), fmaxf(t.z, t.w)));
        }
        for (int s = 16; s > 0; s >>= 1) mx = fmaxf(mx, __shfl_xor(mx, s, 32));
        float sum = 0.0f;
#pragma unroll
        for (int j = 0; j < 32; ++j) { p[j] = __expf(p[j] - mx); sum += p[j]; }
        for (int s = 16; s > 0; s >>= 1) sum += __shfl_xor(sum, s, 32);
        const float inv  = 1.0f / sum;
        const float rowf = (mask[b * 1024 + i0 + r] != 0) ? 0.0f : inv;
        float* gout = attn + (((size_t)(h * 4 + b)) * 1024 + (i0 + r)) * 1024 + lane * 32;
#pragma unroll
        for (int j4 = 0; j4 < 8; ++j4) {
            float4 o;
            o.x = p[j4 * 4 + 0] * rowf * sMF[lane * 32 + j4 * 4 + 0];
            o.y = p[j4 * 4 + 1] * rowf * sMF[lane * 32 + j4 * 4 + 1];
            o.z = p[j4 * 4 + 2] * rowf * sMF[lane * 32 + j4 * 4 + 2];
            o.w = p[j4 * 4 + 3] * rowf * sMF[lane * 32 + j4 * 4 + 3];
            *(float4*)(gout + j4 * 4) = o;                     // attn -> HBM (once)
            *(float4*)(row + lane * 32 + j4 * 4) = o;          // P stays in LDS
        }
    }
    __syncthreads();

    // ---- Stage 4: ctx = P @ V, V chunks (32 k-rows, transposed) double-buffered
    //      in the (now dead) K-tile LDS region ----
    const int vk = tid >> 3, vd0 = (tid & 7) * 8;
    const float* vbase = vb + ((size_t)(b * 1024 + vk)) * 1024 + h * 64 + vd0;
    {
        const float4* p = (const float4*)vbase;       // chunk 0
        float4 v0 = p[0], v1 = p[1];
        float vv[8] = {v0.x, v0.y, v0.z, v0.w, v1.x, v1.y, v1.z, v1.w};
        unsigned short* sVh = (unsigned short*)sKR;
#pragma unroll
        for (int e = 0; e < 8; ++e)
            sVh[(vd0 + e) * 32 + vk] = f2bf(vv[e]);   // d-major, k contiguous
    }
    __syncthreads();

    f32x8 co = {};
    for (int kt = 0; kt < 32; ++kt) {
        const unsigned int* sV = sKR + (kt & 1) * 1024;
        const bool more = (kt < 31);
        float4 v0{}, v1{};
        if (more) {                                   // next chunk: rows += 32
            const float4* p = (const float4*)(vbase + (size_t)(kt + 1) * 32768);
            v0 = p[0]; v1 = p[1];
        }

        Frag a, bf;
#pragma unroll
        for (int v = 0; v < 8; ++v) {
            const int K0 = ((v < 4) ? 0 : 16) + hl * 8 + 2 * (v & 3);
            const float* pp = sS + mr * SSTR + kt * 32 + K0;
            a.u[v] = pack2(pp[0], pp[1]);
        }
        const int dr = nsub * 16 + l16;
        bf.q[0] = *(const U4*)(sV + dr * 16 + hl * 4);
        bf.q[1] = *(const U4*)(sV + dr * 16 + 8 + hl * 4);
        co = __builtin_amdgcn_wmma_f32_16x16x32_bf16(false, a.v, false, bf.v, (short)0, co, false, false);

        if (more) {
            float vv[8] = {v0.x, v0.y, v0.z, v0.w, v1.x, v1.y, v1.z, v1.w};
            unsigned short* sVh = (unsigned short*)(sKR + ((kt + 1) & 1) * 1024);
#pragma unroll
            for (int e = 0; e < 8; ++e)
                sVh[(vd0 + e) * 32 + vk] = f2bf(vv[e]);
        }
        __syncthreads();
    }
    {
        const int dcol = h * 64 + nsub * 16 + l16;
#pragma unroll
        for (int v = 0; v < 8; ++v) {
            const int irow = i0 + msub * 16 + v + 8 * hl;
            ctx[((size_t)(b * 1024 + irow)) * 1024 + dcol] = co[v];
        }
    }
}

// ---------------------------------------------------------------------------
// LayerNorm: one wave32 per row of 1024.
// ---------------------------------------------------------------------------
__launch_bounds__(256)
__global__ void ln_kernel(const float* __restrict__ X, const float* __restrict__ gamma,
                          const float* __restrict__ beta, float* __restrict__ Y)
{
    const int lane = threadIdx.x & 31, w = threadIdx.x >> 5;
    const int row  = blockIdx.x * 8 + w;
    const float* x = X + (size_t)row * 1024;
    float v[32];
    float s = 0.0f, s2 = 0.0f;
#pragma unroll
    for (int j = 0; j < 32; ++j) {
        const float t = x[lane + j * 32];
        v[j] = t; s += t; s2 += t * t;
    }
    for (int m = 16; m > 0; m >>= 1) { s += __shfl_xor(s, m, 32); s2 += __shfl_xor(s2, m, 32); }
    const float mu   = s * (1.0f / 1024.0f);
    const float var  = s2 * (1.0f / 1024.0f) - mu * mu;
    const float rstd = rsqrtf(var + 1e-5f);
    float* y = Y + (size_t)row * 1024;
#pragma unroll
    for (int j = 0; j < 32; ++j) {
        const int c = lane + j * 32;
        y[c] = (v[j] - mu) * rstd * gamma[c] + beta[c];
    }
}

// ---------------------------------------------------------------------------
extern "C" void kernel_launch(void* const* d_in, const int* in_sizes, int n_in,
                              void* d_out, int out_size, void* d_ws, size_t ws_size,
                              hipStream_t stream)
{
    const float* query = (const float*)d_in[0];
    const float* key_  = (const float*)d_in[1];
    const float* value = (const float*)d_in[2];
    const int*   mask  = (const int*)d_in[3];
    const float* wq = (const float*)d_in[4];
    const float* bq = (const float*)d_in[5];
    const float* wk = (const float*)d_in[6];
    const float* bk = (const float*)d_in[7];
    const float* wv = (const float*)d_in[8];
    const float* bv = (const float*)d_in[9];
    const float* wd = (const float*)d_in[10];
    const float* bd = (const float*)d_in[11];
    const float* gamma = (const float*)d_in[12];
    const float* beta  = (const float*)d_in[13];

    float* ws  = (float*)d_ws;
    float* qb  = ws;                  // 4096 x 1024
    float* kb  = qb  + 4194304;
    float* vb  = kb  + 4194304;
    float* ctx = vb  + 4194304;
    float* dn  = ctx + 4194304;

    float* y    = (float*)d_out;      // (4, 1024, 1024)
    float* attn = y + 4194304;        // (64, 1024, 1024)

    const dim3 bb(256, 1, 1);
    const dim3 gg(32, 16, 1);         // 128x64 tiles over 4096x1024
    hipLaunchKernelGGL((gemm_xwt_bias<false>), gg, bb, 0, stream, query, wq, bq, (const float*)nullptr, qb);
    hipLaunchKernelGGL((gemm_xwt_bias<false>), gg, bb, 0, stream, key_,  wk, bk, (const float*)nullptr, kb);
    hipLaunchKernelGGL((gemm_xwt_bias<false>), gg, bb, 0, stream, value, wv, bv, (const float*)nullptr, vb);

    const dim3 ga(32, 16, 4);
    hipLaunchKernelGGL(attn_kernel, ga, bb, 157696 /*bytes dynamic LDS*/, stream, qb, kb, vb, mask, attn, ctx);

    hipLaunchKernelGGL((gemm_xwt_bias<true>), gg, bb, 0, stream, ctx, wd, bd, query, dn);
    hipLaunchKernelGGL(ln_kernel, dim3(512, 1, 1), bb, 0, stream, dn, gamma, beta, y);
}
